// EnhancedGraphSAGE_10050223473232
// MI455X (gfx1250) — matrix operand order repaired and checked by reading, not compiled, not run on hardware
//
#include <hip/hip_runtime.h>
#include <hip/hip_bf16.h>

#define NNODES 50000
#define NEDGES 800000
#define D 256
#define K2 512          // concatenated K: [agg(256) | h(256)]
#define MT 64           // rows per block
#define BN_EPS 1e-5f

typedef _Float16 v16h __attribute__((ext_vector_type(16)));
typedef float    v8f  __attribute__((ext_vector_type(8)));

union BFrag { v16h h; uint4 u[2]; };

// ---------------- utility kernels ----------------
__global__ void fill_zero4(float4* p, int n4) {
    int i = blockIdx.x * blockDim.x + threadIdx.x;
    if (i < n4) p[i] = make_float4(0.f, 0.f, 0.f, 0.f);
}

__global__ void deg_accum(const int* __restrict__ ei, float* __restrict__ deg) {
    int e = blockIdx.x * blockDim.x + threadIdx.x;
    if (e < NEDGES) atomicAdd(deg + ei[NEDGES + e], 1.0f);
}

__global__ void deg_recip(float* deg) {
    int i = blockIdx.x * blockDim.x + threadIdx.x;
    if (i < NNODES) deg[i] = 1.0f / fmaxf(deg[i], 1.0f);
}

// WT layout: [layer][n=256][k=512] f16; k<256 from Wl (applied to agg), k>=256 from Wr
__global__ void convert_weights(const float* __restrict__ Wl0, const float* __restrict__ Wr0,
                                const float* __restrict__ Wl1, const float* __restrict__ Wr1,
                                const float* __restrict__ Wl2, const float* __restrict__ Wr2,
                                _Float16* __restrict__ WT) {
    int idx = blockIdx.x * blockDim.x + threadIdx.x;
    if (idx >= 3 * 256 * 512) return;
    int k = idx & 511;
    int n = (idx >> 9) & 255;
    int layer = idx >> 17;
    const float* Wl = layer == 0 ? Wl0 : (layer == 1 ? Wl1 : Wl2);
    const float* Wr = layer == 0 ? Wr0 : (layer == 1 ? Wr1 : Wr2);
    float v = (k < 256) ? Wl[k * 256 + n] : Wr[(k - 256) * 256 + n];
    WT[idx] = (_Float16)v;
}

// ---------------- edge scatter: agg[dst] += feat[src] ----------------
// one wave per edge; each lane moves 8 contiguous floats (coalesced 1KB row read)
__global__ void scatter_accum(const int* __restrict__ ei,
                              const float* __restrict__ feat,
                              float* __restrict__ agg) {
    int wid  = (blockIdx.x * blockDim.x + threadIdx.x) >> 5;
    int lane = threadIdx.x & 31;
    if (wid >= NEDGES) return;
    int s = ei[wid];
    int d = ei[NEDGES + wid];
    const float4* sp = (const float4*)(feat + (size_t)s * D) + lane * 2;
    float4 v0 = sp[0];
    float4 v1 = sp[1];
    float* dp = agg + (size_t)d * D + lane * 8;
    atomicAdd(dp + 0, v0.x); atomicAdd(dp + 1, v0.y);
    atomicAdd(dp + 2, v0.z); atomicAdd(dp + 3, v0.w);
    atomicAdd(dp + 4, v1.x); atomicAdd(dp + 5, v1.y);
    atomicAdd(dp + 6, v1.z); atomicAdd(dp + 7, v1.w);
}

// ---------------- fused dual-GEMM + bias + BN + ReLU + residual ----------------
// out[m, 0:256] = deg_inv[m]*agg[m,:] @ Wl + h[m,:] @ Wr + b   (then BN/ReLU/residual)
// Block: 256 thr (8 waves), tile 64 rows x 256 cols. Wave w owns cols [32w, 32w+32).
__global__ __launch_bounds__(256)
void sage_gemm_fused(const float* __restrict__ Aagg,
                     const float* __restrict__ Ah,
                     const _Float16* __restrict__ WT,   // [256][512] f16, row n = out col
                     const float* __restrict__ bias,
                     const float* __restrict__ gamma,
                     const float* __restrict__ beta,
                     const float* __restrict__ mean,
                     const float* __restrict__ var,
                     const float* __restrict__ resid,
                     const float* __restrict__ deg_inv,
                     float* __restrict__ out) {
    const int tid  = threadIdx.x;
    const int wave = tid >> 5;
    const int lane = tid & 31;
    const int l15  = lane & 15;
    const int lhi  = lane >> 4;
    const int mbase = blockIdx.x * MT;
    const int cb    = wave * 32;

    v8f acc[4][2] = {};

    int   rowIdx[4];
    float scl[4];
#pragma unroll
    for (int rt = 0; rt < 4; ++rt) {
        int r = mbase + rt * 16 + l15;       // A-fragment row for this lane
        if (r > NNODES - 1) r = NNODES - 1;  // clamp (stores are predicated)
        rowIdx[rt] = r;
        scl[rt]    = deg_inv[r];
    }

#pragma unroll
    for (int phase = 0; phase < 2; ++phase) {          // 0: agg*deg_inv, 1: h
        const float* Aq = phase ? Ah : Aagg;
#pragma unroll
        for (int kk = 0; kk < 8; ++kk) {
            const int ks = kk * 32;                    // K-step within phase
            // B fragments: col n = cb+ct*16+l15, K run = lhi*16 .. +15 (contiguous in WT row)
            BFrag bf[2];
#pragma unroll
            for (int ct = 0; ct < 2; ++ct) {
                int n = cb + ct * 16 + l15;
                const uint4* bp =
                    (const uint4*)(WT + (size_t)n * K2 + phase * 256 + ks + lhi * 16);
                bf[ct].u[0] = bp[0];
                bf[ct].u[1] = bp[1];
            }
            // A fragments: lane holds K = lhi*16 + [0..15] (contiguous, matches B)
            v16h af[4];
#pragma unroll
            for (int rt = 0; rt < 4; ++rt) {
                const float* ap = Aq + (size_t)rowIdx[rt] * D + ks + lhi * 16;
                float4 a0 = *(const float4*)(ap);
                float4 a1 = *(const float4*)(ap + 4);
                float4 a2 = *(const float4*)(ap + 8);
                float4 a3 = *(const float4*)(ap + 12);
                float s = phase ? 1.0f : scl[rt];      // fold mean-normalization into A
                v16h h;
                h[0]  = (_Float16)(a0.x * s); h[1]  = (_Float16)(a0.y * s);
                h[2]  = (_Float16)(a0.z * s); h[3]  = (_Float16)(a0.w * s);
                h[4]  = (_Float16)(a1.x * s); h[5]  = (_Float16)(a1.y * s);
                h[6]  = (_Float16)(a1.z * s); h[7]  = (_Float16)(a1.w * s);
                h[8]  = (_Float16)(a2.x * s); h[9]  = (_Float16)(a2.y * s);
                h[10] = (_Float16)(a2.z * s); h[11] = (_Float16)(a2.w * s);
                h[12] = (_Float16)(a3.x * s); h[13] = (_Float16)(a3.y * s);
                h[14] = (_Float16)(a3.z * s); h[15] = (_Float16)(a3.w * s);
                af[rt] = h;
            }
#pragma unroll
            for (int rt = 0; rt < 4; ++rt)
#pragma unroll
                for (int ct = 0; ct < 2; ++ct)
                    acc[rt][ct] = __builtin_amdgcn_wmma_f32_16x16x32_f16(
                        false, af[rt], false, bf[ct].h, (short)0, acc[rt][ct],
                        false, false);
        }
    }

    // ---------------- epilogue: bias, BN (folded), ReLU, residual ----------------
    const bool hasBN = (gamma != nullptr);
#pragma unroll
    for (int ct = 0; ct < 2; ++ct) {
        int col = cb + ct * 16 + l15;
        float bsc   = bias[col];
        float scale = 1.0f, shift = bsc;
        if (hasBN) {
            float inv = rsqrtf(var[col] + BN_EPS);
            scale = gamma[col] * inv;
            shift = (bsc - mean[col]) * scale + beta[col];
        }
#pragma unroll
        for (int rt = 0; rt < 4; ++rt) {
#pragma unroll
            for (int r = 0; r < 8; ++r) {
                int row = mbase + rt * 16 + r + 8 * lhi;   // C/D layout: M = r + 8*lhi
                if (row >= NNODES) continue;
                float v = acc[rt][ct][r];
                if (hasBN) {
                    v = v * scale + shift;                 // bias + batchnorm
                    v = fmaxf(v, 0.0f);                    // relu
                    v += resid[(size_t)row * D + col];     // residual
                } else {
                    v += bsc;                              // last layer: bias only
                }
                out[(size_t)row * D + col] = v;
            }
        }
    }
}

// ---------------- host launch ----------------
extern "C" void kernel_launch(void* const* d_in, const int* in_sizes, int n_in,
                              void* d_out, int out_size, void* d_ws, size_t ws_size,
                              hipStream_t stream) {
    const float* x   = (const float*)d_in[0];
    const int*   ei  = (const int*)d_in[1];
    const float* Wl0 = (const float*)d_in[2];
    const float* Wr0 = (const float*)d_in[3];
    const float* b0  = (const float*)d_in[4];
    const float* Wl1 = (const float*)d_in[5];
    const float* Wr1 = (const float*)d_in[6];
    const float* b1  = (const float*)d_in[7];
    const float* Wl2 = (const float*)d_in[8];
    const float* Wr2 = (const float*)d_in[9];
    const float* b2  = (const float*)d_in[10];
    const float* gamma0 = (const float*)d_in[11];
    const float* beta0  = (const float*)d_in[12];
    const float* mean0  = (const float*)d_in[13];
    const float* var0   = (const float*)d_in[14];
    const float* gamma1 = (const float*)d_in[15];
    const float* beta1  = (const float*)d_in[16];
    const float* mean1  = (const float*)d_in[17];
    const float* var1   = (const float*)d_in[18];

    // workspace layout (256B aligned)
    char* ws = (char*)d_ws;
    float*    deg = (float*)(ws + 0);                     //   200 KB
    float*    agg = (float*)(ws + 200192);                //  51.2 MB
    float*    hA  = (float*)(ws + 51400192);              //  51.2 MB
    float*    hB  = (float*)(ws + 102600192);             //  51.2 MB
    _Float16* WT  = (_Float16*)(ws + 153800192);          // 768 KB
    float*    outp = (float*)d_out;

    const int ND4 = NNODES * D / 4;                       // 3,200,000 float4s
    const int gemm_blocks = (NNODES + MT - 1) / MT;       // 782
    const int scat_blocks = NEDGES / 8;                   // 100,000 (8 waves/block)

    // one-time: degrees + f16 transposed weights
    fill_zero4<<<(NNODES / 4 + 255) / 256, 256, 0, stream>>>((float4*)deg, NNODES / 4);
    deg_accum<<<(NEDGES + 255) / 256, 256, 0, stream>>>(ei, deg);
    deg_recip<<<(NNODES + 255) / 256, 256, 0, stream>>>(deg);
    convert_weights<<<(3 * 256 * 512 + 255) / 256, 256, 0, stream>>>(
        Wl0, Wr0, Wl1, Wr1, Wl2, Wr2, WT);

    // layer 0: x -> hA  (BN0, ReLU, residual = x)
    fill_zero4<<<(ND4 + 255) / 256, 256, 0, stream>>>((float4*)agg, ND4);
    scatter_accum<<<scat_blocks, 256, 0, stream>>>(ei, x, agg);
    sage_gemm_fused<<<gemm_blocks, 256, 0, stream>>>(
        agg, x, WT, b0, gamma0, beta0, mean0, var0, x, deg, hA);

    // layer 1: hA -> hB (BN1, ReLU, residual = hA)
    fill_zero4<<<(ND4 + 255) / 256, 256, 0, stream>>>((float4*)agg, ND4);
    scatter_accum<<<scat_blocks, 256, 0, stream>>>(ei, hA, agg);
    sage_gemm_fused<<<gemm_blocks, 256, 0, stream>>>(
        agg, hA, WT + 256 * 512, b1, gamma1, beta1, mean1, var1, hA, deg, hB);

    // layer 2: hB -> out (bias only)
    fill_zero4<<<(ND4 + 255) / 256, 256, 0, stream>>>((float4*)agg, ND4);
    scatter_accum<<<scat_blocks, 256, 0, stream>>>(ei, hB, agg);
    sage_gemm_fused<<<gemm_blocks, 256, 0, stream>>>(
        agg, hB, WT + 2 * 256 * 512, b2, nullptr, nullptr, nullptr, nullptr,
        nullptr, deg, outp);
}
